// Qwen2MoeSparseMoeBlock_206158430628
// MI455X (gfx1250) — matrix operand order repaired and checked by reading, not compiled, not run on hardware
//
#include <hip/hip_runtime.h>
#include <hip/hip_bf16.h>
#include <math.h>

// ---------------------------------------------------------------------------
// Qwen2-style sparse MoE block for MI455X (gfx1250), bf16 WMMA + f32 accum.
//   T=2048 tokens, H=2048 hidden, E=16 experts (top-4), I=1408, IS=5632.
// Compute-bound (~0.71 TFLOP/call); weights fit L2 (192MB) so tile re-reads
// are cheap -> big GEMMs on v_wmma_f32_16x16x32_bf16, A-tiles staged to LDS
// by the Tensor Data Mover (tensor_load_to_lds + s_wait_tensorcnt).
// ---------------------------------------------------------------------------

typedef __bf16 bf16;
typedef __attribute__((ext_vector_type(16))) __bf16 v16bf;
typedef __attribute__((ext_vector_type(8)))  float  v8f;
typedef __attribute__((ext_vector_type(8)))  __bf16 bf16x8;
typedef __attribute__((ext_vector_type(4)))  __bf16 bf16x4;
typedef __attribute__((ext_vector_type(4)))  unsigned int uint32x4;
typedef __attribute__((ext_vector_type(8)))  int int32x8;
typedef __attribute__((ext_vector_type(4)))  int int32x4;

#define TILE_M 128
#define TILE_N 64
#define TILE_K 32
#define LDS_STRIDE 40   // 32 + 8 pad (bf16 elems): 64B row + 16B pad
#define THREADS 256     // 8 waves (wave32)

__device__ __forceinline__ float sigmoidf_(float x) {
    return 1.0f / (1.0f + __expf(-x));
}

// ---------------------------------------------------------------------------
// Tensor Data Mover: DMA a [tile_rows x 32] bf16 tile (row stride = lda elems)
// from global into LDS laid out as [row][LDS_STRIDE].  The pad fields insert
// 4 DWORDs (16B) after every 16 DWORDs (64B = one 32-elem bf16 row), which
// reproduces the 40-element padded LDS stride.  Descriptor per ISA 8.3/8.4.
// This toolchain exposes the 6-arg builtin: (g0, g1, g2, g3, g4, cpol).
// ---------------------------------------------------------------------------
__device__ __forceinline__ void tdm_load_a_tile(const bf16* g_tile, const bf16* lds_dst,
                                                unsigned int tensor_d0, unsigned int tensor_d1,
                                                unsigned int stride0_elems, unsigned int tile_rows)
{
    const unsigned long long ga = (unsigned long long)(uintptr_t)g_tile;
    const unsigned int       la = (unsigned int)(uintptr_t)lds_dst; // low 32b = LDS offset

    uint32x4 g0;
    g0[0] = 1u;                                    // count=1 (valid user D#)
    g0[1] = la;                                    // lds_addr
    g0[2] = (unsigned int)ga;                      // global_addr[31:0]
    g0[3] = (unsigned int)(ga >> 32) | (2u << 30); // global_addr[56:32] | type=2

    int32x8 g1;
    g1[0] = (int)((1u << 16)      // data_size = 2 bytes
                | (1u << 20)      // pad_enable
                | (3u << 22)      // pad_interval = 16 DWORDs (one 64B row)
                | (3u << 25));    // pad_amount   = 4 DWORDs (16B)
    g1[1] = (int)((tensor_d0 & 0xFFFFu) << 16);                       // tensor_dim0[15:0]
    g1[2] = (int)(((tensor_d0 >> 16) & 0xFFFFu)
                | ((tensor_d1 & 0xFFFFu) << 16));                     // dim0[31:16] | dim1[15:0]
    g1[3] = (int)(((tensor_d1 >> 16) & 0xFFFFu)
                | ((unsigned int)TILE_K << 16));                      // dim1[31:16] | tile_dim0=32
    g1[4] = (int)tile_rows;                                           // tile_dim1 (tile_dim2=0)
    g1[5] = (int)stride0_elems;                                       // tensor_dim0_stride[31:0]
    g1[6] = 0;                                                        // stride0 hi / stride1 lo
    g1[7] = 0;

    const int32x4 gz4 = {0, 0, 0, 0};            // 2D tensor: groups 2/3 unused
    const int32x8 gz8 = {0, 0, 0, 0, 0, 0, 0, 0};
    __builtin_amdgcn_tensor_load_to_lds(g0, g1, gz4, gz4, gz8, 0);
}

// ---- WMMA fragment loaders (CDNA5 ISA 7.12.2 layouts) ----------------------
// A (16x32 bf16): lane 0-15 row M=lane; K halves split at 8:
// v0..3 -> K {0..7}+half*8, v4..7 -> K {16..23}+half*8.
__device__ __forceinline__ v16bf load_frag_a(const bf16* base, int row_base, int lane) {
    const bf16* r = base + (row_base + (lane & 15)) * LDS_STRIDE;
    const int half = (lane >> 4) & 1;
    v16bf f;
#pragma unroll
    for (int v = 0; v < 8; ++v) {
        const int kb = ((v < 4) ? (2 * v) : (16 + 2 * (v - 4))) + half * 8;
        f[2 * v]     = r[kb];
        f[2 * v + 1] = r[kb + 1];
    }
    return f;
}

// B (32x16 bf16) from an N-major LDS tile (sB[n][k]): same K split as A
// (v0..3 -> K {0..7}+half*8, v4..7 -> K {16..23}+half*8); column N = lane & 15.
__device__ __forceinline__ v16bf load_frag_b(const bf16* base, int n_base, int lane) {
    const bf16* r = base + (n_base + (lane & 15)) * LDS_STRIDE;
    const int half = (lane >> 4) & 1;
    v16bf f;
#pragma unroll
    for (int v = 0; v < 8; ++v) {
        const int kb = ((v < 4) ? (2 * v) : (16 + 2 * (v - 4))) + half * 8;
        f[2 * v]     = r[kb];
        f[2 * v + 1] = r[kb + 1];
    }
    return f;
}

// B tile: 32(K) x 64(N) f32 from a K-major weight matrix, converted to bf16
// and transposed into an N-major LDS tile. 256 thr x 8 elems.
__device__ __forceinline__ void stage_b_f32_T(bf16* sB, const float* __restrict__ W,
                                              int ldb, int col0, int k0, int tid) {
    const int kr = tid >> 3;
    const int cb = (tid & 7) * 8;
    const float* g = W + (size_t)(k0 + kr) * ldb + (col0 + cb);
    __builtin_prefetch(g + (size_t)TILE_K * ldb, 0, 1);   // next K tile
    float4 f0 = *(const float4*)(g);
    float4 f1 = *(const float4*)(g + 4);
    const float tmp[8] = { f0.x, f0.y, f0.z, f0.w, f1.x, f1.y, f1.z, f1.w };
#pragma unroll
    for (int j = 0; j < 8; ++j)
        sB[(size_t)(cb + j) * LDS_STRIDE + kr] = (bf16)tmp[j];
}

// ---------------------------------------------------------------------------
// Kernel: fused gate/up GEMM + SiLU*mul.  act[T,I] = silu(X@Wg) * (X@Wu)
// Wgu is [H, 2*I] f32 (gate = cols [0,I), up = cols [I,2I)).
// grid = (T/128, I/64), block = 256.
// ---------------------------------------------------------------------------
__global__ __launch_bounds__(THREADS) void moe_gate_up_gemm(
    const bf16* __restrict__ Xbf, const float* __restrict__ Wgu,
    bf16* __restrict__ act, int Hdim, int Icols)
{
    __shared__ __align__(16) bf16 sA [TILE_M * LDS_STRIDE];
    __shared__ __align__(16) bf16 sBg[TILE_N * LDS_STRIDE];
    __shared__ __align__(16) bf16 sBu[TILE_N * LDS_STRIDE];

    const int tid  = threadIdx.x;
    const int lane = tid & 31;
    const int wave = tid >> 5;
    const int row0 = blockIdx.x * TILE_M;
    const int col0 = blockIdx.y * TILE_N;
    const int twoI = 2 * Icols;

    const v8f vzero = {};
    v8f accG[4], accU[4];
#pragma unroll
    for (int i = 0; i < 4; ++i) { accG[i] = vzero; accU[i] = vzero; }

    for (int k0 = 0; k0 < Hdim; k0 += TILE_K) {
        if (wave == 0) {   // DMA the A tile via the Tensor Data Mover
            tdm_load_a_tile(Xbf + (size_t)row0 * Hdim + k0, sA,
                            (unsigned int)Hdim, (unsigned int)TILE_M,
                            (unsigned int)Hdim, (unsigned int)TILE_M);
        }
        stage_b_f32_T(sBg, Wgu, twoI, col0,         k0, tid);
        stage_b_f32_T(sBu, Wgu, twoI, Icols + col0, k0, tid);
        if (wave == 0) __builtin_amdgcn_s_wait_tensorcnt(0);
        __syncthreads();   // publishes TDM-written sA + staged sBg/sBu

        const v16bf a = load_frag_a(sA, wave * 16, lane);
#pragma unroll
        for (int nt = 0; nt < 4; ++nt) {
            const v16bf bg = load_frag_b(sBg, nt * 16, lane);
            accG[nt] = __builtin_amdgcn_wmma_f32_16x16x32_bf16(
                false, a, false, bg, (short)0, accG[nt], false, false);
            const v16bf bu = load_frag_b(sBu, nt * 16, lane);
            accU[nt] = __builtin_amdgcn_wmma_f32_16x16x32_bf16(
                false, a, false, bu, (short)0, accU[nt], false, false);
        }
        __syncthreads();
    }

    // Epilogue: C/D layout -> VGPR r holds rows r (lanes 0-15) / r+8 (16-31).
    const int half = lane >> 4;
    const int nlo  = lane & 15;
#pragma unroll
    for (int nt = 0; nt < 4; ++nt) {
#pragma unroll
        for (int r = 0; r < 8; ++r) {
            const int row = row0 + wave * 16 + r + half * 8;
            const int col = col0 + nt * 16 + nlo;
            const float g = accG[nt][r];
            const float u = accU[nt][r];
            act[(size_t)row * Icols + col] = (bf16)(g * sigmoidf_(g) * u);
        }
    }
}

// ---------------------------------------------------------------------------
// Kernel: down GEMM + per-row scaled store/accumulate.
//   out[t,n] (=|+=) rowscale[t] * (act @ Wd)[t,n]
// grid = (T/128, N/64), block = 256.  Stream order serializes accumulators.
// ---------------------------------------------------------------------------
__global__ __launch_bounds__(THREADS) void moe_down_gemm(
    const bf16* __restrict__ act, const float* __restrict__ Wd,
    const float* __restrict__ rowscale, float* __restrict__ out,
    int Kdim, int Ncols, int addMode)
{
    __shared__ __align__(16) bf16 sA[TILE_M * LDS_STRIDE];
    __shared__ __align__(16) bf16 sB[TILE_N * LDS_STRIDE];

    const int tid  = threadIdx.x;
    const int lane = tid & 31;
    const int wave = tid >> 5;
    const int row0 = blockIdx.x * TILE_M;
    const int col0 = blockIdx.y * TILE_N;

    const v8f vzero = {};
    v8f acc[4];
#pragma unroll
    for (int i = 0; i < 4; ++i) acc[i] = vzero;

    for (int k0 = 0; k0 < Kdim; k0 += TILE_K) {
        if (wave == 0) {
            tdm_load_a_tile(act + (size_t)row0 * Kdim + k0, sA,
                            (unsigned int)Kdim, (unsigned int)TILE_M,
                            (unsigned int)Kdim, (unsigned int)TILE_M);
        }
        stage_b_f32_T(sB, Wd, Ncols, col0, k0, tid);
        if (wave == 0) __builtin_amdgcn_s_wait_tensorcnt(0);
        __syncthreads();

        const v16bf a = load_frag_a(sA, wave * 16, lane);
#pragma unroll
        for (int nt = 0; nt < 4; ++nt) {
            const v16bf b = load_frag_b(sB, nt * 16, lane);
            acc[nt] = __builtin_amdgcn_wmma_f32_16x16x32_bf16(
                false, a, false, b, (short)0, acc[nt], false, false);
        }
        __syncthreads();
    }

    const int half = lane >> 4;
    const int nlo  = lane & 15;
#pragma unroll
    for (int r = 0; r < 8; ++r) {
        const int row = row0 + wave * 16 + r + half * 8;
        const float sc = rowscale[row];
#pragma unroll
        for (int nt = 0; nt < 4; ++nt) {
            const int col = col0 + nt * 16 + nlo;
            const size_t o = (size_t)row * Ncols + col;
            const float v = acc[nt][r] * sc;
            out[o] = addMode ? (out[o] + v) : v;
        }
    }
}

// ---------------------------------------------------------------------------
// Router: per token -> 16 expert logits + shared-gate logit, softmax, top-4.
// One wave32 per token (8 tokens / 256-thread block).
// combineT is [E][T] (expert-major) so each down-GEMM reads a contiguous
// per-expert scale vector; rowscale_sh[t] = sigmoid(x . shared_gate_w).
// ---------------------------------------------------------------------------
__global__ __launch_bounds__(THREADS) void moe_router(
    const float* __restrict__ x, const float* __restrict__ gate_w,
    const float* __restrict__ shared_gate_w,
    float* __restrict__ combineT, float* __restrict__ rowscale_sh,
    int T, int H)
{
    const int lane = threadIdx.x & 31;
    const int wave = threadIdx.x >> 5;
    const int t = blockIdx.x * 8 + wave;
    if (t >= T) return;

    const float* xr = x + (size_t)t * H;
    float logits[17];
#pragma unroll
    for (int e = 0; e < 17; ++e) {
        const float* wp = (e < 16) ? (gate_w + (size_t)e * H) : shared_gate_w;
        float p = 0.f;
        for (int h = lane; h < H; h += 32) p += xr[h] * wp[h];
#pragma unroll
        for (int o = 16; o > 0; o >>= 1) p += __shfl_down(p, o, 32);
        logits[e] = p;                       // valid in lane 0
    }

    if (lane == 0) {
        float m = logits[0];
#pragma unroll
        for (int e = 1; e < 16; ++e) m = fmaxf(m, logits[e]);
        float pr[16]; float s = 0.f;
#pragma unroll
        for (int e = 0; e < 16; ++e) { pr[e] = __expf(logits[e] - m); s += pr[e]; }
        const float inv = 1.0f / s;
#pragma unroll
        for (int e = 0; e < 16; ++e) pr[e] *= inv;

        float cw[16];
#pragma unroll
        for (int e = 0; e < 16; ++e) cw[e] = 0.f;
        unsigned taken = 0u;
        for (int k = 0; k < 4; ++k) {          // 4 argmax passes (top-4)
            int best = 0; float bv = -1.f;
#pragma unroll
            for (int e = 0; e < 16; ++e) {
                const bool ok = !((taken >> e) & 1u) && (pr[e] > bv);
                bv   = ok ? pr[e] : bv;
                best = ok ? e : best;
            }
            taken |= (1u << best);
#pragma unroll
            for (int e = 0; e < 16; ++e) if (e == best) cw[e] = bv;
        }
#pragma unroll
        for (int e = 0; e < 16; ++e) combineT[(size_t)e * T + t] = cw[e];
        rowscale_sh[t] = sigmoidf_(logits[16]);
    }
}

// One-time f32 -> bf16 conversion of the token activations.
__global__ void cvt_f32_bf16(const float* __restrict__ in, bf16* __restrict__ out, int n) {
    const int i = (blockIdx.x * blockDim.x + threadIdx.x) * 4;
    if (i + 3 < n) {
        const float4 f = *(const float4*)(in + i);
        bf16x4 o; o[0] = (bf16)f.x; o[1] = (bf16)f.y; o[2] = (bf16)f.z; o[3] = (bf16)f.w;
        *(bf16x4*)(out + i) = o;
    }
}

// ---------------------------------------------------------------------------
extern "C" void kernel_launch(void* const* d_in, const int* in_sizes, int n_in,
                              void* d_out, int out_size, void* d_ws, size_t ws_size,
                              hipStream_t stream) {
    (void)in_sizes; (void)n_in; (void)out_size; (void)ws_size;
    constexpr int T = 2048, H = 2048, E = 16, I = 1408, IS = 5632;

    const float* x             = (const float*)d_in[0];   // [T,H]
    const float* gate_w        = (const float*)d_in[1];   // [E,H]
    const float* w_gate_up     = (const float*)d_in[2];   // [E,H,2I]
    const float* w_down        = (const float*)d_in[3];   // [E,I,H]
    const float* ws_gate_up    = (const float*)d_in[4];   // [H,2*IS]
    const float* ws_down       = (const float*)d_in[5];   // [IS,H]
    const float* shared_gate_w = (const float*)d_in[6];   // [H,1]
    float* out = (float*)d_out;                           // [T,H] f32

    // Workspace layout: X(bf16) | act(bf16, sized for IS) | combineT | rowscale
    char* ws = (char*)d_ws;
    bf16*  x_bf     = (bf16*)ws;
    bf16*  act      = (bf16*)(ws + (size_t)T * H * 2);
    float* combineT = (float*)(ws + (size_t)T * H * 2 + (size_t)T * IS * 2);
    float* rowscale = combineT + (size_t)E * T;

    // 1) X -> bf16 once; reused by all 17 gate/up GEMMs.
    cvt_f32_bf16<<<(T * H) / 1024, 256, 0, stream>>>(x, x_bf, T * H);

    // 2) Router: combine weights (dense [E][T]) + shared sigmoid gate.
    moe_router<<<T / 8, THREADS, 0, stream>>>(x, gate_w, shared_gate_w,
                                              combineT, rowscale, T, H);

    // 3) Shared expert first: stores (initializes) out with sigmoid-gated value.
    moe_gate_up_gemm<<<dim3(T / TILE_M, IS / TILE_N), THREADS, 0, stream>>>(
        x_bf, ws_gate_up, act, H, IS);
    moe_down_gemm<<<dim3(T / TILE_M, H / TILE_N), THREADS, 0, stream>>>(
        act, ws_down, rowscale, out, IS, H, /*addMode=*/0);

    // 4) Routed experts: dense compute, combine-weight masked accumulate.
    //    Stream ordering serializes the read-modify-write on out.
    for (int e = 0; e < E; ++e) {
        moe_gate_up_gemm<<<dim3(T / TILE_M, I / TILE_N), THREADS, 0, stream>>>(
            x_bf, w_gate_up + (size_t)e * H * 2 * I, act, H, I);
        moe_down_gemm<<<dim3(T / TILE_M, H / TILE_N), THREADS, 0, stream>>>(
            act, w_down + (size_t)e * I * H, combineT + (size_t)e * T, out,
            I, H, /*addMode=*/1);
    }
}
